// TRMHasher_33260226740603
// MI455X (gfx1250) — matrix operand order repaired and checked by reading, not compiled, not run on hardware
//
#include <hip/hip_runtime.h>
#include <hip/hip_bf16.h>
#include <stdint.h>

// ---------------------------------------------------------------------------
// CDNA5 (gfx1250) TRMHasher forward.
//  - convs: implicit GEMM, bf16 WMMA (v_wmma_f32_16x16x32_bf16), fused BN+GELU,
//    fully templated geometry -> constant-folded im2col address math
//  - recursion: persistent kernel, 4 WGs x 16 batch rows, LDS-resident state
//  - weights pre-packed into exact wave32 B-fragment layout (2 b128/lane/K-step)
//  - B addresses get an opaque (asm) 0 offset: blocks LICM/CSE of the weight
//    fragment loads across recursion steps (no VGPR blow-up/scratch spills)
//    while preserving global provenance -> global_load_b128 (LOADcnt only,
//    decoupled from the LDS/DScnt stream)
// ---------------------------------------------------------------------------

typedef __attribute__((ext_vector_type(16))) __bf16 v16bf;
typedef __attribute__((ext_vector_type(8)))  float  v8f;

union Frag {
    v16bf v;
    uint4 q[2];
};

__device__ __forceinline__ unsigned short f2bf(float f) {
    unsigned int u = __float_as_uint(f);
    unsigned int r = (u + 0x7FFFu + ((u >> 16) & 1u)) >> 16;   // RNE
    return (unsigned short)r;
}
__device__ __forceinline__ float bf2f(unsigned short h) {
    return __uint_as_float(((unsigned int)h) << 16);
}
__device__ __forceinline__ float gelu_exact(float x) {
    return 0.5f * x * (1.0f + erff(x * 0.70710678118654752440f));
}
__device__ __forceinline__ float silu_f(float x) {
    return x / (1.0f + __expf(-x));
}
__device__ __forceinline__ v8f zero8() {
    v8f c;
#pragma unroll
    for (int i = 0; i < 8; ++i) c[i] = 0.0f;
    return c;
}
// Opaque zero: compiler cannot prove the value, so addresses built with it are
// never provably equal across calls -> loads can't be hoisted/CSE'd out of the
// recursion loop. Unlike laundering the pointer, this keeps the base pointer's
// global address-space provenance (emits global_load_*, not flat_load_*).
__device__ __forceinline__ unsigned opaque_zero() {
    unsigned x = 0u;
    asm volatile("" : "+v"(x));
    return x;
}

// ---------------------------------------------------------------------------
// Weight packing into WMMA B-fragment order:
// block (kb, nb) = 32 lanes x 8 dwords; lane<16: N=nb*16+lane, K=kb*32+2v{,+1} (+8 for v>=4);
// lane>=16: same N set, K += 8. One contiguous 32B load per lane at use time.
// ---------------------------------------------------------------------------
__global__ void pack_conv_w(const float* __restrict__ w, unsigned int* __restrict__ out,
                            int Cout, int Cin, int KH, int KW, int Kreal, int KB, int NB)
{
    int d = blockIdx.x * blockDim.x + threadIdx.x;
    int total = KB * NB * 256;
    if (d >= total) return;
    int blk = d >> 8, rem = d & 255;
    int lane = rem >> 3, vv = rem & 7;
    int kb = blk / NB, nb = blk % NB;
    int n = nb * 16 + (lane & 15);
    int k0 = kb * 32 + ((lane >> 4) << 3) + vv * 2 + ((vv >> 2) << 3);
    int KWC = KW * Cin;
    float f[2] = {0.f, 0.f};
#pragma unroll
    for (int j = 0; j < 2; ++j) {
        int k = k0 + j;
        if (k < Kreal && n < Cout) {
            int kh = k / KWC;
            int r2 = k - kh * KWC;
            int kw = r2 / Cin;
            int c  = r2 - kw * Cin;
            f[j] = w[(((size_t)n * Cin + c) * KH + kh) * KW + kw];   // torch OIHW
        }
    }
    out[d] = (unsigned int)f2bf(f[0]) | ((unsigned int)f2bf(f[1]) << 16);
}

__global__ void pack_dense_w(const float* __restrict__ w, unsigned int* __restrict__ out,
                             int N, int K, int KB, int NB)
{
    int d = blockIdx.x * blockDim.x + threadIdx.x;
    int total = KB * NB * 256;
    if (d >= total) return;
    int blk = d >> 8, rem = d & 255;
    int lane = rem >> 3, vv = rem & 7;
    int kb = blk / NB, nb = blk % NB;
    int n = nb * 16 + (lane & 15);
    int k0 = kb * 32 + ((lane >> 4) << 3) + vv * 2 + ((vv >> 2) << 3);
    float f0 = (n < N) ? w[(size_t)n * K + k0]     : 0.f;   // torch Linear [out,in]
    float f1 = (n < N) ? w[(size_t)n * K + k0 + 1] : 0.f;
    out[d] = (unsigned int)f2bf(f0) | ((unsigned int)f2bf(f1) << 16);
}

// ---------------------------------------------------------------------------
// Generic WMMA tile accumulate: A row-major bf16 (LDS), B packed fragments.
// ---------------------------------------------------------------------------
__device__ __forceinline__ v8f wmma_acc(v8f c,
    const unsigned short* __restrict__ A, int lda,
    const unsigned int* __restrict__ B, int NB, int nb, int kb0, int ksteps)
{
    int ln = threadIdx.x & 31;
    int m  = ln & 15;
    int hi = ln >> 4;
    const unsigned short* arow = A + (size_t)m * lda + hi * 8;
    const unsigned int* Bv = B + ((size_t)kb0 * NB + nb) * 256 + ln * 8 + opaque_zero();
    for (int kb = 0; kb < ksteps; ++kb) {
        Frag a, b;
        const unsigned short* ap = arow + kb * 32;
        a.q[0] = *(const uint4*)(ap);
        a.q[1] = *(const uint4*)(ap + 16);
        const unsigned int* bp = Bv + (size_t)kb * NB * 256;
        b.q[0] = *(const uint4*)(bp);
        b.q[1] = *(const uint4*)(bp + 4);
        c = __builtin_amdgcn_wmma_f32_16x16x32_bf16(false, a.v, false, b.v,
                                                    (short)0, c, false, false);
    }
    return c;
}

// ---------------------------------------------------------------------------
// Implicit-GEMM conv + BN(eval) + exact GELU, NHWC bf16 out.
// Fully templated geometry: all im2col decode divides are by constants
// (strength-reduced), K-loop trip count constant (unroll + s/w pipelining).
// M = 64*HO*WO pixels, N = COUT, K = KH*KW*CIN padded to KB*32.
// F32NCHW=true: gather path from f32 NCHW image (layer 1, Cin=3).
// ---------------------------------------------------------------------------
template<int HI, int WI, int CIN, int HO, int WO, int COUT,
         int KH, int KW, int STRIDE, int PAD, int KREAL, int KB, bool F32NCHW>
__global__ void __launch_bounds__(256)
conv_bn_gelu(const void* __restrict__ src,
             const unsigned int* __restrict__ Bw,
             const float* __restrict__ gamma, const float* __restrict__ beta,
             const float* __restrict__ mean,  const float* __restrict__ var,
             unsigned short* __restrict__ dst)
{
    constexpr int numNt = COUT / 16;
    constexpr int numMt = (64 * HO * WO) / 16;
    constexpr int KWC   = KW * CIN;
    constexpr int HoWo  = HO * WO;

    int wave = blockIdx.x * (blockDim.x >> 5) + (threadIdx.x >> 5);
    if (wave >= numMt * numNt) return;
    int nt = wave % numNt;
    int mt = wave / numNt;
    int ln  = threadIdx.x & 31;
    int hi4 = ln >> 4;

    int m = mt * 16 + (ln & 15);
    int bb = m / HoWo;                 // constant divisor -> mul/shift
    int rm = m - bb * HoWo;
    int ho = rm / WO;
    int wo = rm - ho * WO;
    int hbase = ho * STRIDE - PAD;
    int wbase = wo * STRIDE - PAD;

    const unsigned short* __restrict__ srcH = (const unsigned short*)src;
    const float* __restrict__ srcF = (const float*)src;

    const unsigned int* Bv = Bw + (size_t)nt * 256 + ln * 8 + opaque_zero();

    v8f c = zero8();
    for (int kb = 0; kb < KB; ++kb) {
        Frag a, b;
#pragma unroll
        for (int run = 0; run < 2; ++run) {
            int kk = kb * 32 + hi4 * 8 + run * 16;
            uint4 q; q.x = 0u; q.y = 0u; q.z = 0u; q.w = 0u;
            if (F32NCHW) {
                unsigned short e[8];
#pragma unroll
                for (int j = 0; j < 8; ++j) {
                    int k = kk + j;
                    float val = 0.f;
                    if (k < KREAL) {
                        int kh = k / KWC;
                        int r2 = k - kh * KWC;
                        int kw = r2 / CIN;
                        int cc = r2 - kw * CIN;
                        int hh = hbase + kh, ww = wbase + kw;
                        if (hh >= 0 && hh < HI && ww >= 0 && ww < WI)
                            val = srcF[(((size_t)bb * CIN + cc) * HI + hh) * WI + ww];
                    }
                    e[j] = f2bf(val);
                }
                q.x = (unsigned int)e[0] | ((unsigned int)e[1] << 16);
                q.y = (unsigned int)e[2] | ((unsigned int)e[3] << 16);
                q.z = (unsigned int)e[4] | ((unsigned int)e[5] << 16);
                q.w = (unsigned int)e[6] | ((unsigned int)e[7] << 16);
            } else {
                // CIN%32==0: 8 consecutive K stay in one (kh,kw) cell -> one b128
                int kh = kk / KWC;
                int r2 = kk - kh * KWC;
                int kw = r2 / CIN;
                int cc = r2 - kw * CIN;
                int hh = hbase + kh, ww = wbase + kw;
                if (hh >= 0 && hh < HI && ww >= 0 && ww < WI)
                    q = *(const uint4*)(srcH + (((size_t)bb * HI + hh) * WI + ww) * CIN + cc);
            }
            a.q[run] = q;
        }
        const unsigned int* bp = Bv + (size_t)kb * numNt * 256;
        b.q[0] = *(const uint4*)(bp);
        b.q[1] = *(const uint4*)(bp + 4);
        c = __builtin_amdgcn_wmma_f32_16x16x32_bf16(false, a.v, false, b.v,
                                                    (short)0, c, false, false);
    }
    // fused BN (eval running stats) + exact GELU epilogue, store bf16 NHWC
    int n = nt * 16 + (ln & 15);
    float inv  = gamma[n] * rsqrtf(var[n] + 1e-5f);
    float bias = beta[n] - mean[n] * inv;
#pragma unroll
    for (int r = 0; r < 8; ++r) {
        int M = mt * 16 + r + hi4 * 8;
        float o = c[r] * inv + bias;
        dst[(size_t)M * COUT + n] = f2bf(gelu_exact(o));
    }
}

// AdaptiveAvgPool2d((1,1)) over 7x7, bf16 NHWC -> f32 [64,256]
__global__ void avgpool(const unsigned short* __restrict__ act, float* __restrict__ x)
{
    int b = blockIdx.x;
    int c = threadIdx.x;
    const unsigned short* p = act + (size_t)b * 49 * 256 + c;
    float s = 0.f;
#pragma unroll
    for (int i = 0; i < 49; ++i) s += bf2f(p[i * 256]);
    x[b * 256 + c] = s * (1.0f / 49.0f);
}

// ---------------------------------------------------------------------------
// Recursion helpers (per-workgroup, 16 batch rows, 8 waves)
// ---------------------------------------------------------------------------
__device__ __forceinline__ void rmsnorm16(float (&v)[16][256], const float* __restrict__ w,
                                          unsigned short (&out)[16][256], float (&red)[16])
{
    int tid = threadIdx.x;
    if (tid < 16) red[tid] = 0.f;
    __syncthreads();
    {
        int m  = tid >> 4;
        int c0 = (tid & 15) * 16;
        float s = 0.f;
#pragma unroll
        for (int j = 0; j < 16; ++j) { float t = v[m][c0 + j]; s += t * t; }
        atomicAdd(&red[m], s);
    }
    __syncthreads();
    for (int i = tid; i < 16 * 256; i += 256) {
        int mm = i >> 8, k = i & 255;
        float rr = rsqrtf(red[mm] * (1.0f / 256.0f) + 1e-6f);
        out[mm][k] = f2bf(v[mm][k] * rr * w[k]);
    }
    __syncthreads();
}

__device__ __forceinline__ void rms_final16(float (&v)[16][256], const float* __restrict__ w,
                                            float (&red)[16], float* __restrict__ outG,
                                            unsigned short (&cat)[16][768], int catOff, int m0)
{
    int tid = threadIdx.x;
    if (tid < 16) red[tid] = 0.f;
    __syncthreads();
    {
        int m  = tid >> 4;
        int c0 = (tid & 15) * 16;
        float s = 0.f;
#pragma unroll
        for (int j = 0; j < 16; ++j) { float t = v[m][c0 + j]; s += t * t; }
        atomicAdd(&red[m], s);
    }
    __syncthreads();
    for (int i = tid; i < 16 * 256; i += 256) {
        int mm = i >> 8, k = i & 255;
        float rr  = rsqrtf(red[mm] * (1.0f / 256.0f) + 1e-6f);
        float val = v[mm][k] * rr * w[k];
        outG[(size_t)(m0 + mm) * 256 + k] = val;      // f32 state out
        cat[mm][catOff + k] = f2bf(val);              // bf16 back into cat
    }
    __syncthreads();
}

// One tiny-net application: v = proj(cat slice); 2x TRM block; final rmsnorm
__device__ __forceinline__ void tiny_step(
    unsigned short (&cat)[16][768], float (&vbuf)[16][256],
    unsigned short (&nrm)[16][256], unsigned short (&sbuf)[16][256], float (&red)[16],
    int catOff, int KBproj, const unsigned int* __restrict__ Bproj,
    const unsigned int* const* W1, const unsigned int* const* W2,
    const unsigned int* const* W3, const float* const* NORM,
    const float* __restrict__ fnorm, float* __restrict__ outG, int outCatOff, int m0)
{
    int tid = threadIdx.x;
    int wv  = tid >> 5;
    int ln  = tid & 31;
    int M0r = (ln >> 4) * 8;
    int N   = ln & 15;

    // proj: [16,K] @ [K,256] -> vbuf (N=256 -> 16 tiles over 8 waves)
#pragma unroll
    for (int t = 0; t < 2; ++t) {
        int nb = wv * 2 + t;
        v8f c = zero8();
        c = wmma_acc(c, &cat[0][catOff], 768, Bproj, 16, nb, 0, KBproj);
#pragma unroll
        for (int r = 0; r < 8; ++r) vbuf[M0r + r][nb * 16 + N] = c[r];
    }
    __syncthreads();

    for (int blk = 0; blk < 2; ++blk) {
        rmsnorm16(vbuf, NORM[blk], nrm, red);
        v8f acc0 = zero8(), acc1 = zero8();
        // SwiGLU in 4 chunks of 256 hidden cols; w2 accumulated chunk-wise in regs
        for (int cc = 0; cc < 4; ++cc) {
#pragma unroll
            for (int t = 0; t < 2; ++t) {
                int nbl = wv * 2 + t;
                int nbg = cc * 16 + nbl;
                v8f c1 = zero8(), c3 = zero8();
                c1 = wmma_acc(c1, &nrm[0][0], 256, W1[blk], 64, nbg, 0, 8);
                c3 = wmma_acc(c3, &nrm[0][0], 256, W3[blk], 64, nbg, 0, 8);
#pragma unroll
                for (int r = 0; r < 8; ++r)
                    sbuf[M0r + r][nbl * 16 + N] = f2bf(silu_f(c1[r]) * c3[r]);
            }
            __syncthreads();
            acc0 = wmma_acc(acc0, &sbuf[0][0], 256, W2[blk], 16, wv * 2,     cc * 8, 8);
            acc1 = wmma_acc(acc1, &sbuf[0][0], 256, W2[blk], 16, wv * 2 + 1, cc * 8, 8);
            __syncthreads();
        }
        // residual add
#pragma unroll
        for (int r = 0; r < 8; ++r) {
            vbuf[M0r + r][(wv * 2) * 16 + N]     += acc0[r];
            vbuf[M0r + r][(wv * 2 + 1) * 16 + N] += acc1[r];
        }
        __syncthreads();
    }
    rms_final16(vbuf, fnorm, red, outG, cat, outCatOff, m0);
}

// ---------------------------------------------------------------------------
// Persistent recursion kernel: grid = 4 WGs (16 batch rows each), 256 thr.
// cat = [ xb | yb | zb ] bf16; T=3 cycles x (6 z-updates + 1 y-update); head.
// ---------------------------------------------------------------------------
__global__ void __launch_bounds__(256)
recursion(const float* __restrict__ x,
          const float* __restrict__ y_init, const float* __restrict__ z_init,
          const float* __restrict__ norm0, const float* __restrict__ norm1,
          const float* __restrict__ fnorm,
          const unsigned int* __restrict__ Bxyz, const unsigned int* __restrict__ Byz,
          const unsigned int* __restrict__ Bw1a, const unsigned int* __restrict__ Bw2a,
          const unsigned int* __restrict__ Bw3a,
          const unsigned int* __restrict__ Bw1b, const unsigned int* __restrict__ Bw2b,
          const unsigned int* __restrict__ Bw3b,
          const unsigned int* __restrict__ Bhead,
          float* __restrict__ yOut, float* __restrict__ zOut, float* __restrict__ hOut)
{
    __shared__ __align__(16) unsigned short cat[16][768];   // 24 KB: xb|yb|zb
    __shared__ __align__(16) float          vbuf[16][256];  // 16 KB residual stream
    __shared__ __align__(16) unsigned short nrm[16][256];   //  8 KB rmsnorm out (bf16)
    __shared__ __align__(16) unsigned short sbuf[16][256];  //  8 KB swiglu chunk (bf16)
    __shared__ __align__(16) float          hbuf[16][96];   //  6 KB head out
    __shared__               float          red[16];

    int tid = threadIdx.x;
    int wv  = tid >> 5;
    int ln  = tid & 31;
    int m0  = blockIdx.x * 16;

    for (int i = tid; i < 16 * 256; i += 256) {
        int m = i >> 8, k = i & 255;
        cat[m][k]       = f2bf(x[(size_t)(m0 + m) * 256 + k]);
        cat[m][256 + k] = f2bf(y_init[k]);
        cat[m][512 + k] = f2bf(z_init[k]);
    }
    __syncthreads();

    const unsigned int* W1[2] = {Bw1a, Bw1b};
    const unsigned int* W2[2] = {Bw2a, Bw2b};
    const unsigned int* W3[2] = {Bw3a, Bw3b};
    const float* NORM[2] = {norm0, norm1};

    for (int t = 0; t < 3; ++t) {
        for (int i = 0; i < 6; ++i)
            tiny_step(cat, vbuf, nrm, sbuf, red, 0, 24, Bxyz,
                      W1, W2, W3, NORM, fnorm, zOut, 512, m0);   // z-update
        tiny_step(cat, vbuf, nrm, sbuf, red, 256, 16, Byz,
                  W1, W2, W3, NORM, fnorm, yOut, 256, m0);       // y-update
    }

    // head: h = y @ head.T  (N=96 -> 6 tiles on waves 0..5), then L2-normalize
    if (wv < 6) {
        v8f c = zero8();
        c = wmma_acc(c, &cat[0][256], 768, Bhead, 6, wv, 0, 8);
        int M0r = (ln >> 4) * 8, N = ln & 15;
#pragma unroll
        for (int r = 0; r < 8; ++r) hbuf[M0r + r][wv * 16 + N] = c[r];
    }
    __syncthreads();
    if (tid < 16) {
        float ss = 0.f;
        for (int j = 0; j < 96; ++j) { float t2 = hbuf[tid][j]; ss += t2 * t2; }
        red[tid] = 1.0f / fmaxf(sqrtf(ss), 1e-12f);
    }
    __syncthreads();
    for (int i = tid; i < 16 * 96; i += 256) {
        int mm = i / 96, j = i - mm * 96;
        hOut[(size_t)(m0 + mm) * 96 + j] = hbuf[mm][j] * red[mm];
    }
}

// ---------------------------------------------------------------------------
// Host launcher
// ---------------------------------------------------------------------------
extern "C" void kernel_launch(void* const* d_in, const int* in_sizes, int n_in,
                              void* d_out, int out_size, void* d_ws, size_t ws_size,
                              hipStream_t stream)
{
    (void)in_sizes; (void)n_in; (void)out_size; (void)ws_size;

    const float* img = (const float*)d_in[0];
    const float *cw[4], *cg[4], *cb[4], *cm[4], *cv[4];
    int idx = 1;
    for (int l = 0; l < 4; ++l) {
        cw[l] = (const float*)d_in[idx++];
        cg[l] = (const float*)d_in[idx++];
        cb[l] = (const float*)d_in[idx++];
        cm[l] = (const float*)d_in[idx++];
        cv[l] = (const float*)d_in[idx++];
    }
    const float* norm0 = (const float*)d_in[idx++];
    const float* w1a   = (const float*)d_in[idx++];
    const float* w2a   = (const float*)d_in[idx++];
    const float* w3a   = (const float*)d_in[idx++];
    const float* norm1 = (const float*)d_in[idx++];
    const float* w1b   = (const float*)d_in[idx++];
    const float* w2b   = (const float*)d_in[idx++];
    const float* w3b   = (const float*)d_in[idx++];
    const float* fnorm = (const float*)d_in[idx++];
    const float* pxyz  = (const float*)d_in[idx++];
    const float* pyz   = (const float*)d_in[idx++];
    const float* headw = (const float*)d_in[idx++];
    const float* yini  = (const float*)d_in[idx++];
    const float* zini  = (const float*)d_in[idx++];

    // workspace layout (256B aligned slabs)
    char* ws = (char*)d_ws;
    size_t off = 0;
    auto take = [&](size_t bytes) -> char* {
        char* p = ws + off;
        off += (bytes + 255) & ~(size_t)255;
        return p;
    };
    unsigned short* act1 = (unsigned short*)take((size_t)64 * 112 * 112 * 32 * 2);
    unsigned short* act2 = (unsigned short*)take((size_t)64 * 56 * 56 * 64 * 2);
    unsigned short* act3 = (unsigned short*)take((size_t)64 * 28 * 28 * 128 * 2);
    unsigned short* act4 = (unsigned short*)take((size_t)64 * 49 * 256 * 2);
    float*          xbuf = (float*)take((size_t)64 * 256 * 4);
    unsigned int* pcw1  = (unsigned int*)take((size_t)5  * 2  * 1024);
    unsigned int* pcw2  = (unsigned int*)take((size_t)9  * 4  * 1024);
    unsigned int* pcw3  = (unsigned int*)take((size_t)18 * 8  * 1024);
    unsigned int* pcw4  = (unsigned int*)take((size_t)36 * 16 * 1024);
    unsigned int* pxyzB = (unsigned int*)take((size_t)24 * 16 * 1024);
    unsigned int* pyzB  = (unsigned int*)take((size_t)16 * 16 * 1024);
    unsigned int* pw1a  = (unsigned int*)take((size_t)8  * 64 * 1024);
    unsigned int* pw3a  = (unsigned int*)take((size_t)8  * 64 * 1024);
    unsigned int* pw2a  = (unsigned int*)take((size_t)32 * 16 * 1024);
    unsigned int* pw1b  = (unsigned int*)take((size_t)8  * 64 * 1024);
    unsigned int* pw3b  = (unsigned int*)take((size_t)8  * 64 * 1024);
    unsigned int* pw2b  = (unsigned int*)take((size_t)32 * 16 * 1024);
    unsigned int* pheadB = (unsigned int*)take((size_t)8 * 6 * 1024);

    // pack weights into WMMA B-fragment layout (bf16)
    auto packConv = [&](const float* w, unsigned int* o, int Cout, int Cin,
                        int KH, int KW, int Kreal, int KB, int NB) {
        int total = KB * NB * 256;
        pack_conv_w<<<(total + 255) / 256, 256, 0, stream>>>(w, o, Cout, Cin, KH, KW, Kreal, KB, NB);
    };
    packConv(cw[0], pcw1, 32,  3,   7, 7, 147,  5, 2);
    packConv(cw[1], pcw2, 64,  32,  3, 3, 288,  9, 4);
    packConv(cw[2], pcw3, 128, 64,  3, 3, 576,  18, 8);
    packConv(cw[3], pcw4, 256, 128, 3, 3, 1152, 36, 16);

    auto packDense = [&](const float* w, unsigned int* o, int N, int K) {
        int KB = K / 32, NB = N / 16, total = KB * NB * 256;
        pack_dense_w<<<(total + 255) / 256, 256, 0, stream>>>(w, o, N, K, KB, NB);
    };
    packDense(pxyz, pxyzB, 256, 768);
    packDense(pyz,  pyzB,  256, 512);
    packDense(w1a,  pw1a,  1024, 256);
    packDense(w3a,  pw3a,  1024, 256);
    packDense(w2a,  pw2a,  256, 1024);
    packDense(w1b,  pw1b,  1024, 256);
    packDense(w3b,  pw3b,  1024, 256);
    packDense(w2b,  pw2b,  256, 1024);
    packDense(headw, pheadB, 96, 256);

    // encoder: 4x (conv + BN + GELU), NHWC bf16 activations; templated geometry
    {
        // layer 1: 3->32, k7 s2 p3, gather from f32 NCHW, K 147 padded to 160
        int blocks = (int)((((long long)(64 * 112 * 112 / 16) * (32 / 16)) + 7) / 8);
        conv_bn_gelu<224, 224, 3, 112, 112, 32, 7, 7, 2, 3, 147, 5, true>
            <<<blocks, 256, 0, stream>>>(img, pcw1, cg[0], cb[0], cm[0], cv[0], act1);
    }
    {
        int blocks = (int)((((long long)(64 * 56 * 56 / 16) * (64 / 16)) + 7) / 8);
        conv_bn_gelu<112, 112, 32, 56, 56, 64, 3, 3, 2, 1, 288, 9, false>
            <<<blocks, 256, 0, stream>>>(act1, pcw2, cg[1], cb[1], cm[1], cv[1], act2);
    }
    {
        int blocks = (int)((((long long)(64 * 28 * 28 / 16) * (128 / 16)) + 7) / 8);
        conv_bn_gelu<56, 56, 64, 28, 28, 128, 3, 3, 2, 1, 576, 18, false>
            <<<blocks, 256, 0, stream>>>(act2, pcw3, cg[2], cb[2], cm[2], cv[2], act3);
    }
    {
        int blocks = (int)((((long long)(64 * 7 * 7 / 16) * (256 / 16)) + 7) / 8);
        conv_bn_gelu<28, 28, 128, 7, 7, 256, 3, 3, 4, 1, 1152, 36, false>
            <<<blocks, 256, 0, stream>>>(act3, pcw4, cg[3], cb[3], cm[3], cv[3], act4);
    }

    avgpool<<<64, 256, 0, stream>>>(act4, xbuf);

    // persistent latent recursion + head, outputs (y, z, h) concatenated
    float* out = (float*)d_out;
    recursion<<<4, 256, 0, stream>>>(xbuf, yini, zini, norm0, norm1, fnorm,
        pxyzB, pyzB, pw1a, pw2a, pw3a, pw1b, pw2b, pw3b, pheadB,
        out, out + 64 * 256, out + 2 * 64 * 256);
}
